// Tokenizer_27255862460938
// MI455X (gfx1250) — matrix operand order repaired and checked by reading, not compiled, not run on hardware
//
#include <hip/hip_runtime.h>

#define CC  128
#define GG  2000
#define EE  256
#define BB  20
#define HH  64

#define CHUNK   256                 // elements per chunk (one per thread)
#define CHUNKS  2                   // chunks per block
#define NBLK    ((CC * GG) / (CHUNK * CHUNKS))   // 500

typedef _Float16 v16h __attribute__((ext_vector_type(16)));
typedef _Float16 v8h  __attribute__((ext_vector_type(8)));
typedef float    v8f  __attribute__((ext_vector_type(8)));

// ---------------------------------------------------------------------------
// Main kernel: per-element MLP+softmax (VALU, uniform weights via s_load),
// then probs(16x32,f16) x bin_embed^T(32x16,f16) WMMA with f32 accum,
// fused gene_embed add, write out[c][1+g][e].
// Block = 256 threads = 8 waves; processes 2 chunks of 256 elements.
// ---------------------------------------------------------------------------
__global__ __launch_bounds__(256)
void tok_main_kernel(const float* __restrict__ expr,
                     const float* __restrict__ W1,
                     const float* __restrict__ b1,
                     const float* __restrict__ W2,
                     const float* __restrict__ b2,
                     const float* __restrict__ gene_embed,
                     const float* __restrict__ bin_embed,
                     float* __restrict__ out)
{
    // probs: 16-bit A-matrix layout source (rows of 32 halfs, K padded 20->32)
    __shared__ __align__(16) _Float16 sProbs[CHUNK][32];   // 16 KB
    // bin_embed^T: [n][k], K padded 20->32 -> B fragments are contiguous loads
    __shared__ __align__(16) _Float16 sBinT[EE][32];       // 16 KB

    const unsigned tid        = threadIdx.x;
    const unsigned blockStart = blockIdx.x * (CHUNK * CHUNKS);

    // ---- stage 0: stage bin_embed^T (f16) into LDS, coalesced per k-row ----
    {
        const unsigned n = tid;                  // 0..255 = embedding column
        v8h rows[4];
        #pragma unroll
        for (int q = 0; q < 4; ++q)
            #pragma unroll
            for (int j = 0; j < 8; ++j) {
                const int k = q * 8 + j;
                rows[q][j] = (k < BB) ? (_Float16)bin_embed[k * EE + n]
                                      : (_Float16)0.0f;
            }
        #pragma unroll
        for (int q = 0; q < 4; ++q)
            *(v8h*)&sBinT[n][q * 8] = rows[q];
    }

    const unsigned lane    = tid & 31;
    const unsigned wave    = tid >> 5;
    const unsigned halfSel = lane >> 4;    // 0: lanes 0-15, 1: lanes 16-31
    const unsigned lmod    = lane & 15;

    for (int chunk = 0; chunk < CHUNKS; ++chunk) {
        const unsigned chunkStart = blockStart + chunk * CHUNK;

        __syncthreads();   // sProbs reuse across chunks (also orders stage 0)

        // ---- stage 1: per-element MLP -> softmax -> probs (f16, LDS) -------
        {
            const float e  = expr[chunkStart + tid];
            const float mk = (e != 0.0f) ? 1.0f : 0.0f;

            float acc[BB - 1];
            #pragma unroll
            for (int b = 0; b < BB - 1; ++b) acc[b] = b2[b];

            #pragma unroll 4
            for (int h = 0; h < HH; ++h) {
                float pre = fmaf(e, W1[h], b1[h]);             // uniform -> s_load
                float hv  = (pre > 0.0f) ? pre : 0.01f * pre;  // leaky_relu
                #pragma unroll
                for (int b = 0; b < BB - 1; ++b)
                    acc[b] = fmaf(hv, W2[h * (BB - 1) + b], acc[b]);
            }

            float mx = acc[0];
            #pragma unroll
            for (int b = 1; b < BB - 1; ++b) mx = fmaxf(mx, acc[b]);
            float sum = 0.0f;
            #pragma unroll
            for (int b = 0; b < BB - 1; ++b) { acc[b] = __expf(acc[b] - mx); sum += acc[b]; }
            const float rs = mk / sum;

            // pack probs[0..19] + zero pad to 32 halfs, 4 x ds_store_b128
            v8h p0, p1, p2, p3;
            p0[0] = (_Float16)(1.0f - mk);
            #pragma unroll
            for (int b = 0; b < 7;  ++b) p0[b + 1]  = (_Float16)(acc[b] * rs);
            #pragma unroll
            for (int b = 7; b < 15; ++b) p1[b - 7]  = (_Float16)(acc[b] * rs);
            #pragma unroll
            for (int b = 15; b < 19; ++b) p2[b - 15] = (_Float16)(acc[b] * rs);
            #pragma unroll
            for (int j = 4; j < 8; ++j) p2[j] = (_Float16)0.0f;
            p3 = (v8h)(_Float16)0.0f;
            *(v8h*)&sProbs[tid][0]  = p0;
            *(v8h*)&sProbs[tid][8]  = p1;
            *(v8h*)&sProbs[tid][16] = p2;
            *(v8h*)&sProbs[tid][24] = p3;
        }

        __syncthreads();

        // ---- stage 2: WMMA over E, B fragments shared by 2 M-tiles ---------
        // A fragments for this wave's two 16-element tiles
        v16h A[2];
        unsigned outOff[2][8], geOff[2][8];  // per-lane element offsets (incl. lmod)
        #pragma unroll
        for (int i = 0; i < 2; ++i) {
            const unsigned mt = wave * 2 + i;
            const _Float16* pa = &sProbs[mt * 16 + lmod][0];
            v8h a_lo = *(const v8h*)(pa + halfSel * 8);
            v8h a_hi = *(const v8h*)(pa + 16 + halfSel * 8);
            A[i] = __builtin_shufflevector(a_lo, a_hi,
                                           0,1,2,3,4,5,6,7,8,9,10,11,12,13,14,15);
            #pragma unroll
            for (int v = 0; v < 8; ++v) {
                const unsigned m    = v + halfSel * 8;
                const unsigned elem = chunkStart + mt * 16 + m;
                const unsigned c    = elem / GG;          // unsigned magic-mul div
                const unsigned g    = elem - c * GG;
                outOff[i][v] = (c * (GG + 1) + 1 + g) * EE + lmod;
                geOff[i][v]  = g * EE + lmod;
            }
        }

        #pragma unroll 2
        for (unsigned t = 0; t < EE / 16; ++t) {
            // B fragment: lane = col N=16t+lmod, halfs K=[8*halfSel..+8) and [16+8*halfSel..+8)
            const _Float16* pb = &sBinT[t * 16 + lmod][halfSel * 8];
            v8h b_lo = *(const v8h*)(pb);
            v8h b_hi = *(const v8h*)(pb + 16);
            v16h Bf = __builtin_shufflevector(b_lo, b_hi,
                                              0,1,2,3,4,5,6,7,8,9,10,11,12,13,14,15);

            #pragma unroll
            for (int i = 0; i < 2; ++i) {
                v8f Cz = {0.0f,0.0f,0.0f,0.0f,0.0f,0.0f,0.0f,0.0f};
                v8f D = __builtin_amdgcn_wmma_f32_16x16x32_f16(
                            /*neg_a=*/false, A[i], /*neg_b=*/false, Bf,
                            /*c_mod=*/(short)0, Cz,
                            /*reuse_a=*/false, /*reuse_b=*/false);

                #pragma unroll
                for (int v = 0; v < 8; ++v) {
                    const float ge = gene_embed[geOff[i][v] + t * 16u];
                    out[outOff[i][v] + t * 16u] = D[v] + ge;
                }
            }
        }
    }
}

// ---------------------------------------------------------------------------
// Aux kernel: cond_embed row (position 0 of dim-1) and the float mask plane.
// ---------------------------------------------------------------------------
__global__ __launch_bounds__(256)
void tok_aux_kernel(const int* __restrict__ cond_idx,
                    const float* __restrict__ expr,
                    const float* __restrict__ cond_embed,
                    float* __restrict__ out,
                    float* __restrict__ maskOut)
{
    const unsigned idx = blockIdx.x * 256 + threadIdx.x;

    if (idx < CC * (GG + 1)) {
        unsigned c   = idx / (GG + 1);
        unsigned pos = idx - c * (GG + 1);
        float mv = 0.0f;
        if (pos > 0) mv = (expr[c * GG + (pos - 1)] != 0.0f) ? 1.0f : 0.0f;
        maskOut[idx] = mv;
    }

    if (idx < CC * EE) {
        unsigned c = idx >> 8;     // / 256
        unsigned e = idx & 255;
        out[(c * (GG + 1)) * EE + e] = cond_embed[(unsigned)cond_idx[c] * EE + e];
    }
}

// ---------------------------------------------------------------------------
// Launch
// ---------------------------------------------------------------------------
extern "C" void kernel_launch(void* const* d_in, const int* in_sizes, int n_in,
                              void* d_out, int out_size, void* d_ws, size_t ws_size,
                              hipStream_t stream)
{
    (void)in_sizes; (void)n_in; (void)d_ws; (void)ws_size; (void)out_size;

    const int*   cond_idx   = (const int*)  d_in[0];
    const float* expr       = (const float*)d_in[1];
    const float* W1         = (const float*)d_in[2];
    const float* b1         = (const float*)d_in[3];
    const float* W2         = (const float*)d_in[4];
    const float* b2         = (const float*)d_in[5];
    const float* gene_embed = (const float*)d_in[6];
    const float* bin_embed  = (const float*)d_in[7];
    const float* cond_embed = (const float*)d_in[8];

    float* out     = (float*)d_out;
    float* maskOut = out + (size_t)CC * (GG + 1) * EE;   // 65,568,768 floats

    tok_main_kernel<<<NBLK, 256, 0, stream>>>(
        expr, W1, b1, W2, b2, gene_embed, bin_embed, out);

    const int auxThreads = CC * (GG + 1);                // 256,128 (covers CC*EE too)
    tok_aux_kernel<<<(auxThreads + 255) / 256, 256, 0, stream>>>(
        cond_idx, expr, cond_embed, out, maskOut);
}
